// RecurrentCharLM_14396730376932
// MI455X (gfx1250) — matrix operand order repaired and checked by reading, not compiled, not run on hardware
//
#include <hip/hip_runtime.h>
#include <hip/hip_bf16.h>

typedef _Float16 v16h __attribute__((ext_vector_type(16)));
typedef _Float16 v8h  __attribute__((ext_vector_type(8)));
typedef float    v8f  __attribute__((ext_vector_type(8)));

#define HID   128
#define VOC   96
#define BATCH 64
#define SEQ   32
#define DEPTH 500
#define HPAD  132   // f32 h row stride (pad to dodge bank conflicts)
#define NPAD  136   // f16 normalized row stride

// D = A(16x32 f16) * B(32x16 f16) + C(16x16 f32), one workgroup = 16 batch rows.
// Wave w (of 8) owns output columns [16w,16w+16) and holds W's B-fragments in regs.
__global__ __launch_bounds__(256) void
RecurrentCharLM_gfx1250_kernel(const int* __restrict__ chars,
                               const float* __restrict__ hidden,
                               const float* __restrict__ embed_w,
                               const float* __restrict__ W,
                               const float* __restrict__ bias,
                               const float* __restrict__ pre_scale,
                               const float* __restrict__ post_scale,
                               const float* __restrict__ readout_w,
                               const float* __restrict__ readout_b,
                               float* __restrict__ out)
{
    __shared__ __align__(16) float    h_ls[16 * HPAD];   // resident hidden state (f32)
    __shared__ __align__(16) _Float16 nf16[16 * NPAD];   // normalized h as f16 A-matrix
    __shared__ float inv_rms[16];

    const int t    = threadIdx.x;
    const int blk  = blockIdx.x;        // 4 blocks x 16 rows
    const int lane = t & 31;
    const int wv   = t >> 5;            // wave id 0..7
    const int nl   = lane & 15;         // N (or M) within tile
    const int hi   = lane >> 4;         // lane-half selects K sub-block
    const int n_col = wv * 16 + nl;     // global hidden column 0..127
    const int rt   = t >> 4;            // row for the 256-thread normalize mapping
    const int c0   = (t & 15) * 8;      // 8 contiguous columns per thread
    const int r_s  = 2 * wv + hi;       // row this half-wave reduces for RMS

    // ---- load initial hidden + per-thread constants ----
    #pragma unroll
    for (int i = 0; i < 8; ++i)
        h_ls[rt * HPAD + c0 + i] = hidden[(blk * 16 + rt) * HID + c0 + i];
    float ps8[8], pp8[8];
    #pragma unroll
    for (int i = 0; i < 8; ++i) { ps8[i] = pre_scale[c0 + i]; pp8[i] = post_scale[c0 + i]; }
    const float bias_n = bias[n_col];

    // ---- B-fragments of W (f16), resident in VGPRs for the whole kernel ----
    // B element j of v16h <-> K = kc*32 + hi*16 + j, column = n_col. B[k][n] = W[k*HID+n].
    auto loadW = [&](int kc) {
        v16h r;
        #pragma unroll
        for (int j = 0; j < 16; ++j)
            r[j] = (_Float16)W[(kc * 32 + hi * 16 + j) * HID + n_col];
        return r;
    };
    const v16h wf0 = loadW(0), wf1 = loadW(1), wf2 = loadW(2), wf3 = loadW(3);

    // ---- readout B-fragments (waves 0..5 use them; others load col 0 harmlessly) ----
    const int n_v = (wv < 6) ? (wv * 16 + nl) : 0;   // vocab column
    auto loadR = [&](int kc) {
        v16h r;
        #pragma unroll
        for (int j = 0; j < 16; ++j)
            r[j] = (_Float16)readout_w[n_v * HID + kc * 32 + hi * 16 + j];
        return r;
    };
    const v16h rf0 = loadR(0), rf1 = loadR(1), rf2 = loadR(2), rf3 = loadR(3);
    const float rbias = readout_b[n_v];

    // A-fragment: lane holds row M = nl; contiguous K order matching B ->
    // elem j: K = kc*32 + hi*16 + j.
    auto loadA = [&](int kc) {
        const int base = nl * NPAD + kc * 32 + hi * 16;
        v8h lo = *(const v8h*)(nf16 + base);
        v8h hv = *(const v8h*)(nf16 + base + 8);
        return __builtin_shufflevector(lo, hv, 0,1,2,3,4,5,6,7,8,9,10,11,12,13,14,15);
    };

    __syncthreads();

    #pragma unroll 1
    for (int s = 0; s < SEQ; ++s) {
        // per-timestep embed injection: e[m][n_col] for the 8 rows this lane updates
        float e_reg[8];
        #pragma unroll
        for (int v = 0; v < 8; ++v) {
            const int m  = v + 8 * hi;
            const int ch = chars[(blk * 16 + m) * SEQ + s];
            e_reg[v] = embed_w[ch * HID + n_col];
        }

        #pragma unroll 1
        for (int d = 0; d < DEPTH; ++d) {
            // --- row RMS (each half-wave reduces one row via shuffles) ---
            float p = 0.f;
            #pragma unroll
            for (int i = 0; i < 8; ++i) {
                float x = h_ls[r_s * HPAD + nl * 8 + i];
                p += x * x;
            }
            p += __shfl_xor(p, 1, 32); p += __shfl_xor(p, 2, 32);
            p += __shfl_xor(p, 4, 32); p += __shfl_xor(p, 8, 32);
            if (nl == 0) inv_rms[r_s] = rsqrtf(p * (1.0f / HID) + 1e-6f);
            __syncthreads();

            // --- normalize -> f16 A-matrix in LDS ---
            const float ir = inv_rms[rt];
            #pragma unroll
            for (int i = 0; i < 8; ++i)
                nf16[rt * NPAD + c0 + i] =
                    (_Float16)(h_ls[rt * HPAD + c0 + i] * ir * ps8[i]);
            __syncthreads();

            // --- 16x128 @ 128x16 tile: 4 chained WMMAs, f32 accumulate ---
            v8f c = {0.f,0.f,0.f,0.f,0.f,0.f,0.f,0.f};
            c = __builtin_amdgcn_wmma_f32_16x16x32_f16(false, loadA(0), false, wf0, (short)0, c, false, false);
            c = __builtin_amdgcn_wmma_f32_16x16x32_f16(false, loadA(1), false, wf1, (short)0, c, false, false);
            c = __builtin_amdgcn_wmma_f32_16x16x32_f16(false, loadA(2), false, wf2, (short)0, c, false, false);
            c = __builtin_amdgcn_wmma_f32_16x16x32_f16(false, loadA(3), false, wf3, (short)0, c, false, false);

            // --- fused residual ReLU update: h += relu(y + e + b) ---
            #pragma unroll
            for (int v = 0; v < 8; ++v) {
                const int m = v + 8 * hi;             // D-matrix row for VGPR v
                const float y = c[v] + e_reg[v] + bias_n;
                h_ls[m * HPAD + n_col] += (y > 0.f ? y : 0.f);
            }
            __syncthreads();
        }

        // --- post-RMSNorm: h <- norm(h)*post_scale, also as f16 for readout WMMA ---
        {
            float p = 0.f;
            #pragma unroll
            for (int i = 0; i < 8; ++i) {
                float x = h_ls[r_s * HPAD + nl * 8 + i];
                p += x * x;
            }
            p += __shfl_xor(p, 1, 32); p += __shfl_xor(p, 2, 32);
            p += __shfl_xor(p, 4, 32); p += __shfl_xor(p, 8, 32);
            if (nl == 0) inv_rms[r_s] = rsqrtf(p * (1.0f / HID) + 1e-6f);
        }
        __syncthreads();
        {
            const float ir = inv_rms[rt];
            #pragma unroll
            for (int i = 0; i < 8; ++i) {
                const float v_ = h_ls[rt * HPAD + c0 + i] * ir * pp8[i];
                h_ls[rt * HPAD + c0 + i] = v_;
                nf16[rt * NPAD + c0 + i] = (_Float16)v_;
            }
        }
        __syncthreads();

        // --- logits = h_post @ readout_w^T + readout_b (waves 0..5, N=96) ---
        if (wv < 6) {
            v8f c = {0.f,0.f,0.f,0.f,0.f,0.f,0.f,0.f};
            c = __builtin_amdgcn_wmma_f32_16x16x32_f16(false, loadA(0), false, rf0, (short)0, c, false, false);
            c = __builtin_amdgcn_wmma_f32_16x16x32_f16(false, loadA(1), false, rf1, (short)0, c, false, false);
            c = __builtin_amdgcn_wmma_f32_16x16x32_f16(false, loadA(2), false, rf2, (short)0, c, false, false);
            c = __builtin_amdgcn_wmma_f32_16x16x32_f16(false, loadA(3), false, rf3, (short)0, c, false, false);
            #pragma unroll
            for (int v = 0; v < 8; ++v) {
                const int m = v + 8 * hi;
                out[((blk * 16 + m) * SEQ + s) * VOC + n_v] = c[v] + rbias;
            }
        }
        __syncthreads();
    }

    // --- h_final (post-normed hidden of last step), appended after logits ---
    #pragma unroll
    for (int i = 0; i < 8; ++i)
        out[BATCH * SEQ * VOC + (blk * 16 + rt) * HID + c0 + i] =
            h_ls[rt * HPAD + c0 + i];
}

extern "C" void kernel_launch(void* const* d_in, const int* in_sizes, int n_in,
                              void* d_out, int out_size, void* d_ws, size_t ws_size,
                              hipStream_t stream) {
    (void)in_sizes; (void)n_in; (void)d_ws; (void)ws_size; (void)out_size;
    const int*   chars      = (const int*)  d_in[0];
    const float* hidden     = (const float*)d_in[1];
    const float* embed_w    = (const float*)d_in[2];
    const float* W          = (const float*)d_in[3];
    const float* b          = (const float*)d_in[4];
    const float* pre_scale  = (const float*)d_in[5];
    const float* post_scale = (const float*)d_in[6];
    const float* readout_w  = (const float*)d_in[7];
    const float* readout_b  = (const float*)d_in[8];
    float* out = (float*)d_out;

    RecurrentCharLM_gfx1250_kernel<<<dim3(4), dim3(256), 0, stream>>>(
        chars, hidden, embed_w, W, b, pre_scale, post_scale,
        readout_w, readout_b, out);
}
